// ExpertLayer_56281251447212
// MI455X (gfx1250) — matrix operand order repaired and checked by reading, not compiled, not run on hardware
//
#include <hip/hip_runtime.h>
#include <hip/hip_bf16.h>
#include <math.h>

// ---------------- model constants ----------------
#define D_MODEL   1024
#define D_FF      4096
#define N_EXPERTS 8
#define N_TOKENS  8192          // 4 * 2048
#define M_TILE    32            // tokens per FFN block (2 x 16-row m-subtiles)
#define FC        256           // d_ff chunk per fused iteration
#define MAX_TILES (N_TOKENS / M_TILE)   // 256
#define OUT_ELEMS (N_TOKENS * D_MODEL)  // 8388608

// ---------------- WMMA types ----------------
typedef __attribute__((ext_vector_type(16))) __bf16 v16bf;
typedef __attribute__((ext_vector_type(8)))  float  v8f;
typedef __attribute__((ext_vector_type(4)))  unsigned int v4u;

struct A16 { v4u lo, hi; };     // 32 bytes -> bit_cast to v16bf

__device__ __forceinline__ unsigned short f2bf(float f) {
  unsigned u = __float_as_uint(f);
  unsigned r = ((u >> 16) & 1u) + 0x7FFFu;       // round-to-nearest-even
  return (unsigned short)((u + r) >> 16);
}

__device__ __forceinline__ v16bf load_afrag(const unsigned short* ap) {
  A16 t;
  t.lo = *(const v4u*)ap;
  t.hi = *(const v4u*)(ap + 8);
  return __builtin_bit_cast(v16bf, t);
}

// ---------------- init ----------------
__global__ void moe_zero_kernel(float* out, int n, int* cnt) {
  int i = blockIdx.x * blockDim.x + threadIdx.x;
  for (; i < n; i += gridDim.x * blockDim.x) out[i] = 0.0f;
  if (blockIdx.x == 0 && threadIdx.x < N_EXPERTS) cnt[threadIdx.x] = 0;
}

// ---------------- x fp32 -> bf16 (row-major kept) ----------------
__global__ void moe_cvt_x(const float* __restrict__ x, unsigned short* __restrict__ xbf) {
  int i = blockIdx.x * blockDim.x + threadIdx.x;   // index in groups of 4
  float4 v = ((const float4*)x)[i];
  ushort4 o;
  o.x = f2bf(v.x); o.y = f2bf(v.y); o.z = f2bf(v.z); o.w = f2bf(v.w);
  ((ushort4*)xbf)[i] = o;
}

// --------- W [E][R][C] fp32 -> Wt [E][C][R] bf16 (transpose + convert) ---------
__global__ void moe_transpose_cvt(const float* __restrict__ src,
                                  unsigned short* __restrict__ dst,
                                  int R, int C) {
  __shared__ float tile[32][33];
  const float* s = src + (size_t)blockIdx.z * R * C;
  unsigned short* d = dst + (size_t)blockIdx.z * R * C;
  int c0 = blockIdx.x * 32, r0 = blockIdx.y * 32;
  for (int j = threadIdx.y; j < 32; j += 8)
    tile[j][threadIdx.x] = s[(size_t)(r0 + j) * C + c0 + threadIdx.x];
  __syncthreads();
  for (int j = threadIdx.y; j < 32; j += 8)
    d[(size_t)(c0 + j) * R + r0 + threadIdx.x] = f2bf(tile[threadIdx.x][j]);
}

// ---------------- gating: softmax, top-2, routing lists ----------------
__global__ void moe_gate_kernel(const float* __restrict__ x,
                                const float* __restrict__ gw,
                                int* __restrict__ tok, float* __restrict__ wass,
                                int* __restrict__ cnt, float* __restrict__ gprobs) {
  const int lane = threadIdx.x & 31;
  const int wave = threadIdx.x >> 5;
  const int t = blockIdx.x * 8 + wave;            // one wave32 per token
  float acc[N_EXPERTS];
#pragma unroll
  for (int e = 0; e < N_EXPERTS; ++e) acc[e] = 0.0f;
  const float* xr = x + (size_t)t * D_MODEL;
  for (int i = lane; i < D_MODEL; i += 32) {
    float xv = xr[i];
#pragma unroll
    for (int e = 0; e < N_EXPERTS; ++e) acc[e] += xv * gw[e * D_MODEL + i];
  }
#pragma unroll
  for (int off = 16; off > 0; off >>= 1)
#pragma unroll
    for (int e = 0; e < N_EXPERTS; ++e) acc[e] += __shfl_xor(acc[e], off, 32);

  if (lane == 0) {
    float mx = acc[0];
#pragma unroll
    for (int e = 1; e < N_EXPERTS; ++e) mx = fmaxf(mx, acc[e]);
    float p[N_EXPERTS], s = 0.0f;
#pragma unroll
    for (int e = 0; e < N_EXPERTS; ++e) { p[e] = expf(acc[e] - mx); s += p[e]; }
    float inv = 1.0f / s;
#pragma unroll
    for (int e = 0; e < N_EXPERTS; ++e) { p[e] *= inv; gprobs[t * N_EXPERTS + e] = p[e]; }
    int i1 = 0; float p1 = p[0];
#pragma unroll
    for (int e = 1; e < N_EXPERTS; ++e) if (p[e] > p1) { p1 = p[e]; i1 = e; }
    int i2 = -1; float p2 = -1.0f;
#pragma unroll
    for (int e = 0; e < N_EXPERTS; ++e) if (e != i1 && p[e] > p2) { p2 = p[e]; i2 = e; }
    float wsum = p1 + p2;
    int pos1 = atomicAdd(&cnt[i1], 1);
    tok[i1 * N_TOKENS + pos1]  = t;
    wass[i1 * N_TOKENS + pos1] = p1 / wsum;
    int pos2 = atomicAdd(&cnt[i2], 1);
    tok[i2 * N_TOKENS + pos2]  = t;
    wass[i2 * N_TOKENS + pos2] = p2 / wsum;
  }
}

// ---------------- fused expert FFN: GEMM1 -> GELU -> GEMM2 -> scatter ----------------
// 32-token tile: every B-fragment load feeds 2 WMMAs (two 16-row m-subtiles),
// halving expert-weight L2 traffic per routed token vs a 16-token tile.
__global__ void __launch_bounds__(256)
moe_ffn_kernel(const unsigned short* __restrict__ xbf,
               const unsigned short* __restrict__ w1t,   // [E][D_FF][D_MODEL] bf16
               const unsigned short* __restrict__ w2t,   // [E][D_MODEL][D_FF] bf16
               const float* __restrict__ b1, const float* __restrict__ b2,
               const int* __restrict__ tok, const float* __restrict__ wass,
               const int* __restrict__ cnt, float* __restrict__ out) {
  __shared__ unsigned short xt[M_TILE * D_MODEL];   // 64 KB gathered x tile
  __shared__ unsigned short hb[M_TILE * FC];        // 16 KB GELU'd h chunk
  __shared__ int   tks[M_TILE];
  __shared__ float wts[M_TILE];

  const int e    = blockIdx.x / MAX_TILES;
  const int tile = blockIdx.x % MAX_TILES;
  const int count = cnt[e];
  if (tile * M_TILE >= count) return;               // uniform early-exit
  const int nrows = min(M_TILE, count - tile * M_TILE);

  const int tid = threadIdx.x;
  if (tid < M_TILE) {
    int idx = tile * M_TILE + tid;
    bool live = tid < nrows;
    tks[tid] = tok[e * N_TOKENS + (live ? idx : tile * M_TILE)];
    wts[tid] = live ? wass[e * N_TOKENS + idx] : 0.0f;
  }
  __syncthreads();

  // gather 32 token rows of bf16 x into LDS (16B vectors)
  for (int j = tid; j < M_TILE * (D_MODEL / 8); j += 256) {
    int row = j >> 7;           // D_MODEL/8 = 128 v4u per row
    int col = j & 127;
    ((v4u*)xt)[row * 128 + col] = ((const v4u*)(xbf + (size_t)tks[row] * D_MODEL))[col];
  }
  __syncthreads();

  const int lane = tid & 31;
  const int wave = tid >> 5;
  const int half = lane >> 4;          // 0 | 1
  const int l16  = lane & 15;
  const int koA  = half * 16;          // A-frag: 16 contiguous K at koA
  const int koB  = half * 16;          // B-frag: 16 contiguous K at koB
  const unsigned short* w1e = w1t + (size_t)e * D_FF * D_MODEL;
  const unsigned short* w2e = w2t + (size_t)e * D_MODEL * D_FF;

  const v8f zero8 = {0.f, 0.f, 0.f, 0.f, 0.f, 0.f, 0.f, 0.f};
  v8f c2[2][8];                        // persistent out accumulators (2 subtiles x 8 n-tiles)
#pragma unroll
  for (int st = 0; st < 2; ++st)
#pragma unroll
    for (int t2 = 0; t2 < 8; ++t2) c2[st][t2] = zero8;

  for (int ch = 0; ch < D_FF / FC; ++ch) {          // 16 chunks of 256 ff cols
    // ---- GEMM1: this wave computes ff cols [ch*FC + wave*32, +32) over K=1024 ----
    v8f c1[2][2];
#pragma unroll
    for (int st = 0; st < 2; ++st) { c1[st][0] = zero8; c1[st][1] = zero8; }
    const int ffw = ch * FC + wave * 32;
    for (int ks = 0; ks < D_MODEL / 32; ++ks) {
      const int kb = ks * 32;
      v16bf a0 = load_afrag(xt + l16 * D_MODEL + kb + koA);
      v16bf a1 = load_afrag(xt + (16 + l16) * D_MODEL + kb + koA);
#pragma unroll
      for (int nt = 0; nt < 2; ++nt) {
        const unsigned short* bp =
            w1e + (size_t)(ffw + nt * 16 + l16) * D_MODEL + kb + koB;
        v16bf bv = *(const v16bf*)bp;               // 32B contiguous along K
        c1[0][nt] = __builtin_amdgcn_wmma_f32_16x16x32_bf16(
            false, a0, false, bv, (short)0, c1[0][nt], false, false);
        c1[1][nt] = __builtin_amdgcn_wmma_f32_16x16x32_bf16(
            false, a1, false, bv, (short)0, c1[1][nt], false, false);
      }
    }
    // ---- bias + exact GELU -> bf16 h chunk in LDS ----
#pragma unroll
    for (int nt = 0; nt < 2; ++nt) {
      const int ffcol = ffw + nt * 16 + l16;
      const float bias = b1[e * D_FF + ffcol];
#pragma unroll
      for (int st = 0; st < 2; ++st) {
#pragma unroll
        for (int r = 0; r < 8; ++r) {
          int m = st * 16 + r + half * 8;           // C-frag row
          float v = c1[st][nt][r] + bias;
          float g = 0.5f * v * (1.0f + erff(v * 0.70710678118654752f));
          hb[m * FC + (ffcol - ch * FC)] = f2bf(g);
        }
      }
    }
    __syncthreads();
    // ---- GEMM2: K = this 256-wide ff chunk, N = 1024 out cols (8 tiles/wave) ----
    for (int kk = 0; kk < FC / 32; ++kk) {
      const int kb = kk * 32;
      v16bf a0 = load_afrag(hb + l16 * FC + kb + koA);
      v16bf a1 = load_afrag(hb + (16 + l16) * FC + kb + koA);
#pragma unroll
      for (int tt = 0; tt < 8; ++tt) {
        const int oc = (wave * 8 + tt) * 16 + l16;
        const unsigned short* bp =
            w2e + (size_t)oc * D_FF + ch * FC + kb + koB;
        v16bf bv = *(const v16bf*)bp;
        c2[0][tt] = __builtin_amdgcn_wmma_f32_16x16x32_bf16(
            false, a0, false, bv, (short)0, c2[0][tt], false, false);
        c2[1][tt] = __builtin_amdgcn_wmma_f32_16x16x32_bf16(
            false, a1, false, bv, (short)0, c2[1][tt], false, false);
      }
    }
    __syncthreads();
  }

  // ---- epilogue: bias, gate weight, scatter-add (2 commutative adds/token) ----
#pragma unroll
  for (int tt = 0; tt < 8; ++tt) {
    const int oc = (wave * 8 + tt) * 16 + l16;
    const float bias = b2[e * D_MODEL + oc];
#pragma unroll
    for (int st = 0; st < 2; ++st) {
#pragma unroll
      for (int r = 0; r < 8; ++r) {
        int m = st * 16 + r + half * 8;
        if (m < nrows) {
          float v = (c2[st][tt][r] + bias) * wts[m];
          atomicAdd(out + (size_t)tks[m] * D_MODEL + oc, v);
        }
      }
    }
  }
}

// ---------------- aux loss (deterministic fixed-order reduction) ----------------
__global__ void moe_finalize_kernel(const float* __restrict__ gprobs, float* out_aux) {
  __shared__ float red[256];
  float ps[N_EXPERTS], cs[N_EXPERTS];
#pragma unroll
  for (int e = 0; e < N_EXPERTS; ++e) { ps[e] = 0.f; cs[e] = 0.f; }
  for (int t = threadIdx.x; t < N_TOKENS; t += 256) {
#pragma unroll
    for (int e = 0; e < N_EXPERTS; ++e) {
      float p = gprobs[t * N_EXPERTS + e];
      ps[e] += p;
      cs[e] += (p > 0.0f) ? 1.0f : 0.0f;
    }
  }
  float aux = 0.0f;
  for (int e = 0; e < N_EXPERTS; ++e) {
    red[threadIdx.x] = ps[e];
    __syncthreads();
    float sp = 0.f;
    if (threadIdx.x == 0) for (int i = 0; i < 256; ++i) sp += red[i];
    __syncthreads();
    red[threadIdx.x] = cs[e];
    __syncthreads();
    if (threadIdx.x == 0) {
      float sc = 0.f;
      for (int i = 0; i < 256; ++i) sc += red[i];
      aux += (sp / (float)N_TOKENS) * (sc / (float)N_TOKENS);
    }
    __syncthreads();
  }
  if (threadIdx.x == 0) *out_aux = aux * (float)N_EXPERTS * 0.01f;
}

// ---------------- launch ----------------
extern "C" void kernel_launch(void* const* d_in, const int* in_sizes, int n_in,
                              void* d_out, int out_size, void* d_ws, size_t ws_size,
                              hipStream_t stream) {
  const float* x  = (const float*)d_in[0];
  const float* gw = (const float*)d_in[1];
  const float* W1 = (const float*)d_in[2];
  const float* b1 = (const float*)d_in[3];
  const float* W2 = (const float*)d_in[4];
  const float* b2 = (const float*)d_in[5];
  float* out = (float*)d_out;

  // workspace carve-up (~152 MB)
  unsigned short* xbf = (unsigned short*)d_ws;                        // 16 MB
  unsigned short* w1t = xbf + (size_t)N_TOKENS * D_MODEL;             // 64 MB
  unsigned short* w2t = w1t + (size_t)N_EXPERTS * D_FF * D_MODEL;     // 64 MB
  int*   tok  = (int*)(w2t + (size_t)N_EXPERTS * D_MODEL * D_FF);     // 256 KB
  float* wass = (float*)(tok + N_EXPERTS * N_TOKENS);                 // 256 KB
  float* gpr  = wass + N_EXPERTS * N_TOKENS;                          // 256 KB
  int*   cnt  = (int*)(gpr + (size_t)N_TOKENS * N_EXPERTS);           // 32 B

  moe_zero_kernel<<<2048, 256, 0, stream>>>(out, OUT_ELEMS + 1, cnt);
  moe_cvt_x<<<(OUT_ELEMS / 4) / 256, 256, 0, stream>>>(x, xbf);
  dim3 tb(32, 8);
  moe_transpose_cvt<<<dim3(D_FF / 32, D_MODEL / 32, N_EXPERTS), tb, 0, stream>>>(
      W1, w1t, D_MODEL, D_FF);
  moe_transpose_cvt<<<dim3(D_MODEL / 32, D_FF / 32, N_EXPERTS), tb, 0, stream>>>(
      W2, w2t, D_FF, D_MODEL);
  moe_gate_kernel<<<N_TOKENS / 8, 256, 0, stream>>>(x, gw, tok, wass, cnt, gpr);
  moe_ffn_kernel<<<N_EXPERTS * MAX_TILES, 256, 0, stream>>>(
      xbf, w1t, w2t, b1, b2, tok, wass, cnt, out);
  moe_finalize_kernel<<<1, 256, 0, stream>>>(gpr, out + OUT_ELEMS);
}
